// GQARepeat_7344394076579
// MI455X (gfx1250) — matrix operand order repaired and checked by reading, not compiled, not run on hardware
//
#include <hip/hip_runtime.h>
#include <hip/hip_bf16.h>

// ---------------------------------------------------------------------------
// GQA attention block for MI455X (gfx1250), bf16 WMMA path.
//   B=2, S=2048, HID=2048, H=32, G=8, D=64, QPG=4
// Pipeline:
//   1. convert x -> bf16; convert+transpose Wq/Wk/Wv/Wo -> bf16 [N,K]
//   2. Q/K/V projections: register-blocked 64x64-per-wave bf16 WMMA GEMM
//   3. transpose V -> Vt[b,g,d,t] (bf16) for contiguous B-fragments in P*V
//   4. flash attention: 32-query tile per wave, v_wmma_f32_16x16x32_bf16 for
//      QK^T and P*V (K/V fragments reused across 2 query sub-tiles),
//      online softmax, P transposed via per-wave LDS tile
//   5. output projection: same WMMA GEMM -> f32 d_out
// ---------------------------------------------------------------------------

typedef __attribute__((ext_vector_type(16))) __bf16        v16bf;
typedef __attribute__((ext_vector_type(8)))  float         v8f;
typedef __attribute__((ext_vector_type(4)))  unsigned int  u32x4;

#define Bq   2
#define Sq   2048
#define HIDq 2048
#define Hq   32
#define Gq   8
#define Dq   64
#define KVq  (Gq * Dq)   // 512
#define Mq   (Bq * Sq)   // 4096 tokens

__device__ __forceinline__ unsigned short f2bf(float f) {
  union { float f; unsigned u; } v; v.f = f;
  unsigned r = v.u + 0x7FFFu + ((v.u >> 16) & 1u);   // round-to-nearest-even
  return (unsigned short)(r >> 16);
}

union BF16Frag {
  v16bf v;
  u32x4 q[2];
  unsigned short h[16];
};

// B-fragment (32x16, K x N): lane = column n, elements e -> K = 16*half + e.
__device__ __forceinline__ v16bf load_b_frag(const unsigned short* p) {
  BF16Frag f;
  f.q[0] = *(const u32x4*)(p);
  f.q[1] = *(const u32x4*)(p + 8);
  return f.v;
}

// A-fragment (16x32, M x K): lane = row m, elements e -> K = k0+16*half+e.
__device__ __forceinline__ v16bf load_a_frag(const unsigned short* row, int k0, int half) {
  BF16Frag f;
  f.q[0] = *(const u32x4*)(row + k0 + 16 * half);
  f.q[1] = *(const u32x4*)(row + k0 + 16 * half + 8);
  return f.v;
}

__device__ __forceinline__ v8f wmma_bf16(v16bf a, v16bf b, v8f c) {
  return __builtin_amdgcn_wmma_f32_16x16x32_bf16(
      /*neg_a=*/false, a, /*neg_b=*/false, b,
      /*c_mod=*/(short)0, c, /*reuse_a=*/false, /*reuse_b=*/false);
}

// ------------------------------ helper kernels ------------------------------

__global__ __launch_bounds__(256) void cvt_f32_bf16(const float* __restrict__ in,
                                                    unsigned short* __restrict__ out,
                                                    int n) {
  int i = blockIdx.x * 256 + threadIdx.x;
  if (i < n) out[i] = f2bf(in[i]);
}

// W[K,N] f32 -> Wt[N,K] bf16
__global__ __launch_bounds__(256) void wtrans_bf16(const float* __restrict__ W,
                                                   unsigned short* __restrict__ Wt,
                                                   int K, int N) {
  int i = blockIdx.x * 256 + threadIdx.x;
  if (i >= K * N) return;
  int n = i / K, k = i % K;
  Wt[i] = f2bf(W[(size_t)k * N + n]);
}

// V[b,t,g*D+d] bf16 -> Vt[b,g,d,t] bf16
__global__ __launch_bounds__(256) void vtrans(const unsigned short* __restrict__ V,
                                              unsigned short* __restrict__ Vt) {
  int i = blockIdx.x * 256 + threadIdx.x;
  int total = Bq * Gq * Dq * Sq;
  if (i >= total) return;
  int t = i % Sq;
  int d = (i / Sq) % Dq;
  int g = (i / (Sq * Dq)) % Gq;
  int b = i / (Sq * Dq * Gq);
  Vt[i] = V[((size_t)(b * Sq + t)) * KVq + g * Dq + d];
}

// ------------------------------ WMMA GEMM -----------------------------------
// C[M,N] = A[M,K](bf16) @ Bt[N,K]^T(bf16) + bias[N].
// 64x64 output block per wave: 4 A-frags x 4 B-frags -> 16 WMMAs per k-step,
// 1 b128 load per WMMA (vs 4 for a 16x16-per-wave scheme).
template <bool OUT_F32>
__global__ __launch_bounds__(256) void gemm_wmma(const unsigned short* __restrict__ A,
                                                 const unsigned short* __restrict__ Bt,
                                                 const float* __restrict__ bias,
                                                 void* __restrict__ Cout,
                                                 int M, int N, int K) {
  int wid = (blockIdx.x * 256 + threadIdx.x) >> 5;
  int tilesN = N >> 6;                     // 64-wide
  int tilesM = M >> 6;                     // 64-tall
  if (wid >= tilesM * tilesN) return;
  int tm = wid / tilesN, tn = wid % tilesN;
  int lane = threadIdx.x & 31;
  int half = lane >> 4, lm = lane & 15;

  const unsigned short* arow[4];
  const unsigned short* brow[4];
#pragma unroll
  for (int i = 0; i < 4; ++i) {
    arow[i] = A  + (size_t)(tm * 64 + i * 16 + lm) * K;
    brow[i] = Bt + (size_t)(tn * 64 + i * 16 + lm) * K;
  }

  v8f acc[4][4];
#pragma unroll
  for (int i = 0; i < 4; ++i)
#pragma unroll
    for (int j = 0; j < 4; ++j) acc[i][j] = (v8f){};

  for (int k0 = 0; k0 < K; k0 += 32) {
    // pull the next k-tiles toward the WGP while this step's WMMAs run
    if (k0 + 64 < K) {
#pragma unroll
      for (int i = 0; i < 4; ++i) {
        __builtin_prefetch(arow[i] + k0 + 64, 0, 3);
        __builtin_prefetch(brow[i] + k0 + 64, 0, 3);
      }
    }
    v16bf a[4], b[4];
#pragma unroll
    for (int i = 0; i < 4; ++i) a[i] = load_a_frag(arow[i], k0, half);
#pragma unroll
    for (int j = 0; j < 4; ++j) b[j] = load_b_frag(brow[j] + k0 + 16 * half);
#pragma unroll
    for (int i = 0; i < 4; ++i)
#pragma unroll
      for (int j = 0; j < 4; ++j)
        acc[i][j] = wmma_bf16(a[i], b[j], acc[i][j]);
  }

#pragma unroll
  for (int j = 0; j < 4; ++j) {
    int col = tn * 64 + j * 16 + lm;
    float bv = bias[col];
#pragma unroll
    for (int i = 0; i < 4; ++i) {
      size_t rbase = (size_t)(tm * 64 + i * 16 + 8 * half);
      if (OUT_F32) {
        float* C = (float*)Cout;
#pragma unroll
        for (int r = 0; r < 8; ++r)
          C[(rbase + r) * N + col] = acc[i][j][r] + bv;
      } else {
        unsigned short* C = (unsigned short*)Cout;
#pragma unroll
        for (int r = 0; r < 8; ++r)
          C[(rbase + r) * N + col] = f2bf(acc[i][j][r] + bv);
      }
    }
  }
}

// --------------------------- flash attention --------------------------------
// 32-query tile per wave (2 x 16-row sub-tiles sharing K/V fragments);
// 8 waves / block; grid sized exactly so __syncthreads stays uniform.
__global__ __launch_bounds__(256) void attn_wmma(const unsigned short* __restrict__ Q,
                                                 const unsigned short* __restrict__ Kc,
                                                 const unsigned short* __restrict__ Vt,
                                                 unsigned short* __restrict__ O) {
  __shared__ unsigned short P[8][32][32];   // per-wave P-tile (32 queries x 32 keys)

  int wv   = threadIdx.x >> 5;
  int w    = blockIdx.x * 8 + wv;           // global wave tile id
  int qt   = w & 63;                        // S/32 = 64 query tiles
  int h    = (w >> 6) & 31;
  int b    = w >> 11;
  int g    = h >> 2;                        // QPG = 4
  int lane = threadIdx.x & 31;
  int half = lane >> 4, lm = lane & 15;

  // Q fragments: [u = query sub-tile][c = head-dim chunk]
  v16bf qa[2][2];
#pragma unroll
  for (int u = 0; u < 2; ++u) {
    const unsigned short* qrow =
        Q + ((size_t)(b * Sq + qt * 32 + u * 16 + lm)) * HIDq + h * Dq;
    qa[u][0] = load_a_frag(qrow, 0,  half);
    qa[u][1] = load_a_frag(qrow, 32, half);
  }

  const unsigned short* kbase = Kc + (size_t)b * Sq * KVq + g * Dq;
  const unsigned short* vbase = Vt + ((size_t)(b * Gq + g) * Dq) * Sq;

  v8f o[2][4];
#pragma unroll
  for (int u = 0; u < 2; ++u)
#pragma unroll
    for (int f = 0; f < 4; ++f) o[u][f] = (v8f){};
  float mrun[2][8], lrun[2][8];
#pragma unroll
  for (int u = 0; u < 2; ++u)
#pragma unroll
    for (int j = 0; j < 8; ++j) { mrun[u][j] = -1e30f; lrun[u][j] = 0.f; }

  const float scale = 0.125f;               // 1/sqrt(64)

  for (int t0 = 0; t0 < Sq; t0 += 32) {
    // ---- scores: S = Q @ K^T (32 queries x 32 keys) ----
    v16bf kb[2][2];                         // [v = key sub-tile][c = dim chunk]
#pragma unroll
    for (int v = 0; v < 2; ++v) {
      const unsigned short* kr = kbase + (size_t)(t0 + v * 16 + lm) * KVq;
      kb[v][0] = load_b_frag(kr + 16 * half);
      kb[v][1] = load_b_frag(kr + 32 + 16 * half);
    }
    v8f s[2][2];
#pragma unroll
    for (int u = 0; u < 2; ++u)
#pragma unroll
      for (int v = 0; v < 2; ++v) {
        v8f t = wmma_bf16(qa[u][0], kb[v][0], (v8f){});
        s[u][v] = wmma_bf16(qa[u][1], kb[v][1], t);
      }

    // ---- online softmax over the 32 keys, both query sub-tiles ----
#pragma unroll
    for (int u = 0; u < 2; ++u) {
#pragma unroll
      for (int j = 0; j < 8; ++j) {
        float v0 = s[u][0][j] * scale, v1 = s[u][1][j] * scale;
        float mt = fmaxf(v0, v1);
        mt = fmaxf(mt, __shfl_xor(mt, 1));
        mt = fmaxf(mt, __shfl_xor(mt, 2));
        mt = fmaxf(mt, __shfl_xor(mt, 4));
        mt = fmaxf(mt, __shfl_xor(mt, 8));
        float mnew = fmaxf(mrun[u][j], mt);
        float corr = __expf(mrun[u][j] - mnew);
        float p0 = __expf(v0 - mnew);
        float p1 = __expf(v1 - mnew);
        float rs = p0 + p1;
        rs += __shfl_xor(rs, 1);
        rs += __shfl_xor(rs, 2);
        rs += __shfl_xor(rs, 4);
        rs += __shfl_xor(rs, 8);
        lrun[u][j] = lrun[u][j] * corr + rs;
        mrun[u][j] = mnew;
        o[u][0][j] *= corr; o[u][1][j] *= corr;
        o[u][2][j] *= corr; o[u][3][j] *= corr;
        P[wv][u * 16 + j + 8 * half][lm]      = f2bf(p0);
        P[wv][u * 16 + j + 8 * half][lm + 16] = f2bf(p1);
      }
    }
    __syncthreads();   // uniform across block; orders LDS writes vs reads

    // ---- reload P as A fragments (one per query sub-tile) ----
    BF16Frag pa[2];
#pragma unroll
    for (int u = 0; u < 2; ++u) {
      pa[u].q[0] = *(const u32x4*)(&P[wv][u * 16 + lm][16 * half]);
      pa[u].q[1] = *(const u32x4*)(&P[wv][u * 16 + lm][16 * half + 8]);
    }

    // ---- O += P @ V; Vt fragments shared by both query sub-tiles ----
#pragma unroll
    for (int f = 0; f < 4; ++f) {
      v16bf vb = load_b_frag(vbase + (size_t)(f * 16 + lm) * Sq + t0 + 16 * half);
      o[0][f] = wmma_bf16(pa[0].v, vb, o[0][f]);
      o[1][f] = wmma_bf16(pa[1].v, vb, o[1][f]);
    }
    __syncthreads();
  }

  // ---- normalize and store O (bf16) ----
#pragma unroll
  for (int u = 0; u < 2; ++u)
#pragma unroll
    for (int j = 0; j < 8; ++j) {
      float il = 1.f / lrun[u][j];
      unsigned short* orow =
          O + ((size_t)(b * Sq + qt * 32 + u * 16 + j + 8 * half)) * HIDq + h * Dq;
      orow[lm]      = f2bf(o[u][0][j] * il);
      orow[16 + lm] = f2bf(o[u][1][j] * il);
      orow[32 + lm] = f2bf(o[u][2][j] * il);
      orow[48 + lm] = f2bf(o[u][3][j] * il);
    }
}

// ------------------------------- launcher -----------------------------------

extern "C" void kernel_launch(void* const* d_in, const int* in_sizes, int n_in,
                              void* d_out, int out_size, void* d_ws, size_t ws_size,
                              hipStream_t stream) {
  const float* x  = (const float*)d_in[0];
  const float* Wq = (const float*)d_in[1];
  const float* bq = (const float*)d_in[2];
  const float* Wk = (const float*)d_in[3];
  const float* bk = (const float*)d_in[4];
  const float* Wv = (const float*)d_in[5];
  const float* bv = (const float*)d_in[6];
  const float* Wo = (const float*)d_in[7];
  const float* bo = (const float*)d_in[8];
  float* out = (float*)d_out;

  // workspace carve-up (bf16 buffers), ~84 MB total
  char* ws = (char*)d_ws;
  auto take = [&](size_t bytes) { char* p = ws; ws += (bytes + 255) & ~size_t(255); return (unsigned short*)p; };
  unsigned short* xb  = take((size_t)Mq * HIDq * 2);
  unsigned short* Wqt = take((size_t)HIDq * HIDq * 2);
  unsigned short* Wkt = take((size_t)HIDq * KVq * 2);
  unsigned short* Wvt = take((size_t)HIDq * KVq * 2);
  unsigned short* Wot = take((size_t)HIDq * HIDq * 2);
  unsigned short* Qb  = take((size_t)Mq * HIDq * 2);
  unsigned short* Kb  = take((size_t)Mq * KVq * 2);
  unsigned short* Vb  = take((size_t)Mq * KVq * 2);
  unsigned short* Vtb = take((size_t)Mq * KVq * 2);
  unsigned short* Ob  = take((size_t)Mq * HIDq * 2);

  auto cdiv = [](int a, int b) { return (a + b - 1) / b; };

  // 1. conversions / transposes
  int nx = Mq * HIDq;
  cvt_f32_bf16<<<cdiv(nx, 256), 256, 0, stream>>>(x, xb, nx);
  wtrans_bf16<<<cdiv(HIDq * HIDq, 256), 256, 0, stream>>>(Wq, Wqt, HIDq, HIDq);
  wtrans_bf16<<<cdiv(HIDq * KVq, 256), 256, 0, stream>>>(Wk, Wkt, HIDq, KVq);
  wtrans_bf16<<<cdiv(HIDq * KVq, 256), 256, 0, stream>>>(Wv, Wvt, HIDq, KVq);
  wtrans_bf16<<<cdiv(HIDq * HIDq, 256), 256, 0, stream>>>(Wo, Wot, HIDq, HIDq);

  // 2. projections (bf16 out); 64x64 tiles per wave
  {
    int tiles = (Mq / 64) * (HIDq / 64);           // 64*32 = 2048 waves
    gemm_wmma<false><<<cdiv(tiles * 32, 256), 256, 0, stream>>>(xb, Wqt, bq, Qb, Mq, HIDq, HIDq);
  }
  {
    int tiles = (Mq / 64) * (KVq / 64);            // 64*8 = 512 waves
    gemm_wmma<false><<<cdiv(tiles * 32, 256), 256, 0, stream>>>(xb, Wkt, bk, Kb, Mq, KVq, HIDq);
    gemm_wmma<false><<<cdiv(tiles * 32, 256), 256, 0, stream>>>(xb, Wvt, bv, Vb, Mq, KVq, HIDq);
  }

  // 3. V transpose for contiguous P*V B-fragments
  vtrans<<<cdiv(Mq * KVq, 256), 256, 0, stream>>>(Vb, Vtb);

  // 4. flash attention: B*H*(S/32) = 4096 wave tiles, 8 waves/block -> 512 blocks
  attn_wmma<<<(Bq * Hq * (Sq / 32)) / 8, 256, 0, stream>>>(Qb, Kb, Vtb, Ob);

  // 5. output projection (f32 out)
  {
    int tiles = (Mq / 64) * (HIDq / 64);
    gemm_wmma<true><<<cdiv(tiles * 32, 256), 256, 0, stream>>>(Ob, Wot, bo, out, Mq, HIDq, HIDq);
  }
}
